// SelfAttention_86346022519409
// MI455X (gfx1250) — matrix operand order repaired and checked by reading, not compiled, not run on hardware
//
#include <hip/hip_runtime.h>

// ---------------------------------------------------------------------------
// Self-attention for (B=8, C=256, H=W=64) on MI455X (gfx1250, wave32, WMMA).
//
// - All GEMMs on v_wmma_f32_16x16x32_bf16 (K=32 == D exactly).
// - Fused flash attention (online softmax): the 537 MB softmax matrix is
//   never materialized.
// - V blocks (16 KB / 32 keys) staged into LDS by the Tensor Data Mover
//   (tensor_load_to_lds, TENSORcnt, double-buffered), shared by all 8 waves
//   of the workgroup: cuts dominant L2 traffic 8x vs. per-wave streaming.
// ---------------------------------------------------------------------------

typedef __attribute__((ext_vector_type(16))) __bf16 v16bf;
typedef __attribute__((ext_vector_type(8)))  __bf16 v8bf;
typedef __attribute__((ext_vector_type(8)))  float  v8f;
typedef __attribute__((ext_vector_type(4)))  unsigned int u32x4;
typedef __attribute__((ext_vector_type(8)))  int i32x8;
typedef __attribute__((ext_vector_type(4)))  int i32x4;

#define BATCH 8
#define CCH   256
#define NSEQ  4096
#define DCH   32
#define VROW  80   // LDS bytes per V row: 64 data + 16 TDM pad (bank spread)

__device__ __forceinline__ v8f wmma_bf16(v16bf a, v16bf b, v8f c) {
  return __builtin_amdgcn_wmma_f32_16x16x32_bf16(false, a, false, b, (short)0, c,
                                                 false, false);
}

// A tile as loaded here (contiguous k; B loads use the same order): lane L holds row M = L&15.
// lanes 0-15: K=0..15 (elems 0-15); lanes 16-31: K=16..31.
__device__ __forceinline__ v16bf load_a_tile(const __bf16* base, int stride, int lane) {
  const int row = lane & 15;
  const int kb  = (lane & 16) ? 16 : 0;
  const __bf16* p = base + row * stride;
  v8bf lo = *(const v8bf*)(p + kb);
  v8bf hi = *(const v8bf*)(p + kb + 8);
  v16bf r;
#pragma unroll
  for (int j = 0; j < 8; ++j) { r[j] = lo[j]; r[j + 8] = hi[j]; }
  return r;
}

// C/D 16x16 f32 layout: elem r <-> row M = r + ((lane&16)?8:0), col = lane&15.
__device__ __forceinline__ float half_reduce_max(float v) {
#pragma unroll
  for (int m = 1; m <= 8; m <<= 1) v = fmaxf(v, __shfl_xor(v, m, 32));
  return v;
}
__device__ __forceinline__ float half_reduce_add(float v) {
#pragma unroll
  for (int m = 1; m <= 8; m <<= 1) v += __shfl_xor(v, m, 32);
  return v;
}

// TDM: load a (256 x 32) bf16 tile (rows = channels e, contiguous dim = keys m)
// out of the (256 x 4096) V tensor into LDS, padding each 64 B row with 16 B.
// D# packing per CDNA5 ISA.  6-arg builtin
// (clang-23 / therock-10.0 signature): (g0, g1, g2, g3, g4, cpol).
__device__ __forceinline__ void tdm_load_v(const __bf16* gsrc, unsigned lds_off) {
  const unsigned long long ga = (unsigned long long)(size_t)gsrc;
  u32x4 g0;
  g0[0] = 1u;                                              // count=1, user desc
  g0[1] = lds_off;                                         // lds_addr
  g0[2] = (unsigned)(ga & 0xffffffffu);                    // global_addr[31:0]
  g0[3] = (unsigned)((ga >> 32) & 0x01ffffffu) | 0x80000000u; // addr[56:32] | type=2
  i32x8 g1;
  g1[0] = 0x06D10000;  // data_size=2B | pad_enable | pad_interval=16dw | pad_amount=4dw
  g1[1] = 0x10000000;  // tensor_dim0 = 4096 (bits 63:48)
  g1[2] = 0x01000000;  // tensor_dim1 = 256  (bits 95:80)
  g1[3] = 0x00200000;  // tile_dim0   = 32   (bits 127:112)
  g1[4] = 256;         // tile_dim1   = 256  (bits 143:128)
  g1[5] = 4096;        // tensor_dim0_stride = 4096 (bits 191:160)
  g1[6] = 0;
  g1[7] = 0;
  const i32x4 gz4 = {0, 0, 0, 0};
  const i32x8 gz8 = {0, 0, 0, 0, 0, 0, 0, 0};
  __builtin_amdgcn_tensor_load_to_lds(g0, g1, gz4, gz4, gz8, 0);
}

// ---------------------------------------------------------------------------
// Kernel 1: weights fp32 -> bf16
// ---------------------------------------------------------------------------
__global__ void convert_w_kernel(const float* __restrict__ Wq,
                                 const float* __restrict__ Wk,
                                 const float* __restrict__ Wv,
                                 __bf16* __restrict__ Wqb,
                                 __bf16* __restrict__ Wkb,
                                 __bf16* __restrict__ Wvb) {
  int i = blockIdx.x * blockDim.x + threadIdx.x;
  if (i < DCH * CCH) { Wqb[i] = (__bf16)Wq[i]; Wkb[i] = (__bf16)Wk[i]; }
  if (i < CCH * CCH) { Wvb[i] = (__bf16)Wv[i]; }
}

// ---------------------------------------------------------------------------
// Kernel 2: xT[b][n][c] = bf16(x[b][c][n])   (LDS tiled transpose, coalesced)
// ---------------------------------------------------------------------------
__global__ __launch_bounds__(256) void transpose_conv_kernel(
    const float* __restrict__ x, __bf16* __restrict__ xT) {
  __shared__ float tile[32][33];
  const int tx = threadIdx.x, ty = threadIdx.y;          // block (32, 8)
  const int n0 = blockIdx.x * 32, c0 = blockIdx.y * 32, b = blockIdx.z;
  const float* xb = x + (size_t)b * CCH * NSEQ;
  __bf16* xTb = xT + (size_t)b * NSEQ * CCH;
#pragma unroll
  for (int i = 0; i < 4; ++i) {
    const int c = c0 + ty + i * 8;
    tile[ty + i * 8][tx] = xb[(size_t)c * NSEQ + n0 + tx];
  }
  __syncthreads();
#pragma unroll
  for (int i = 0; i < 4; ++i) {
    const int n = n0 + ty + i * 8;
    xTb[(size_t)n * CCH + c0 + tx] = (__bf16)tile[tx][ty + i * 8];
  }
}

// ---------------------------------------------------------------------------
// Kernel 3: q/k projections (one wave -> one 16(n) x 16(d) tile, K=C=256).
// ---------------------------------------------------------------------------
__global__ __launch_bounds__(256) void proj_qk_kernel(
    const __bf16* __restrict__ xT, const __bf16* __restrict__ Wqb,
    const __bf16* __restrict__ Wkb, const float* __restrict__ bq,
    const float* __restrict__ bk, __bf16* __restrict__ qT,
    __bf16* __restrict__ kT) {
  const int tid = threadIdx.x, w = tid >> 5, lane = tid & 31;
  const int colN = lane & 15, half = (lane >> 4) & 1, kb16 = half << 4;
  const int gw    = blockIdx.x * 8 + w;   // 8192 waves total
  const int which = gw & 1;               // 0=q, 1=k
  const int dtile = (gw >> 1) & 1;        // D=32 -> 2 tiles
  const int ntile = (gw >> 2) & 255;
  const int b     = gw >> 10;
  const __bf16* W    = which ? Wkb : Wqb;
  const float*  bias = which ? bk : bq;
  __bf16*       dst  = which ? kT : qT;
  const __bf16* xTb  = xT + ((size_t)b * NSEQ + ntile * 16) * CCH;

  v8f acc = {};
#pragma unroll
  for (int cb = 0; cb < CCH; cb += 32) {
    v16bf a  = load_a_tile(xTb + cb, CCH, lane);
    v16bf bm = *(const v16bf*)(W + (dtile * 16 + colN) * CCH + cb + kb16);
    acc = wmma_bf16(a, bm, acc);
  }
  const float bia = bias[dtile * 16 + colN];
  __bf16* dp = dst + ((size_t)b * NSEQ + ntile * 16 + (half << 3)) * DCH +
               dtile * 16 + colN;
#pragma unroll
  for (int r = 0; r < 8; ++r) dp[(size_t)r * DCH] = (__bf16)(acc[r] + bia);
}

// ---------------------------------------------------------------------------
// Kernel 4: v projection (one wave -> one 16(e) x 16(n) tile, K=256).
// ---------------------------------------------------------------------------
__global__ __launch_bounds__(256) void proj_v_kernel(
    const __bf16* __restrict__ xT, const __bf16* __restrict__ Wvb,
    const float* __restrict__ bv, __bf16* __restrict__ vbuf) {
  const int tid = threadIdx.x, w = tid >> 5, lane = tid & 31;
  const int colN = lane & 15, half = (lane >> 4) & 1, kb16 = half << 4;
  const int gw    = blockIdx.x * 8 + w;   // 32768 waves total
  const int ntile = gw & 255;
  const int etile = (gw >> 8) & 15;
  const int b     = gw >> 12;
  const __bf16* Arow = Wvb + (size_t)(etile * 16) * CCH;
  const __bf16* xTb  = xT + (size_t)b * NSEQ * CCH;

  v8f acc = {};
#pragma unroll
  for (int cb = 0; cb < CCH; cb += 32) {
    v16bf a  = load_a_tile(Arow + cb, CCH, lane);
    v16bf bm = *(const v16bf*)(xTb + (size_t)(ntile * 16 + colN) * CCH + cb + kb16);
    acc = wmma_bf16(a, bm, acc);
  }
  __bf16* dp = vbuf + ((size_t)b * CCH + etile * 16 + (half << 3)) * NSEQ +
               ntile * 16 + colN;
#pragma unroll
  for (int r = 0; r < 8; ++r)
    dp[(size_t)r * NSEQ] = (__bf16)(acc[r] + bv[etile * 16 + (half << 3) + r]);
}

// ---------------------------------------------------------------------------
// Kernel 5: fused flash attention.  8 waves/WG, one 16-query tile per wave,
// 128 key blocks of 32.  V blocks staged once per WG by the TDM into LDS
// (double-buffered), published with s_wait_tensorcnt + workgroup barriers.
// ---------------------------------------------------------------------------
__global__ __launch_bounds__(256) void attn_kernel(
    const __bf16* __restrict__ qT, const __bf16* __restrict__ kT,
    const __bf16* __restrict__ vbuf, const float* __restrict__ x,
    const float* __restrict__ gamma, float* __restrict__ out) {
  __shared__ __align__(128) unsigned char Vlds[2][CCH * VROW];  // 2 x 20 KB
  __shared__ __bf16 Plds[8][16][32];                            // 8 KB
  const int tid = threadIdx.x, w = tid >> 5, lane = tid & 31;
  const int colN = lane & 15, half = (lane >> 4) & 1, kb16 = half << 4;
  const int b = blockIdx.y;
  const int qtile = blockIdx.x * 8 + w;
  const int qbase = qtile * 16;

  const __bf16* qT_b = qT + (size_t)b * NSEQ * DCH;
  const __bf16* kT_b = kT + (size_t)b * NSEQ * DCH;
  const __bf16* v_b  = vbuf + (size_t)b * CCH * NSEQ;
  const float*  x_b  = x + (size_t)b * CCH * NSEQ;
  float*        o_b  = out + (size_t)b * CCH * NSEQ;
  const float g = gamma[0];

  const unsigned ldsv0 = (unsigned)(size_t)(void*)&Vlds[0][0];
  const unsigned ldsv1 = (unsigned)(size_t)(void*)&Vlds[1][0];

  const v16bf a_q = load_a_tile(qT_b + (size_t)qbase * DCH, DCH, lane);

  v8f O[16];
  float m_r[8], l_r[8];
#pragma unroll
  for (int t = 0; t < 16; ++t) O[t] = v8f{};
#pragma unroll
  for (int r = 0; r < 8; ++r) { m_r[r] = -1e30f; l_r[r] = 0.0f; }

  // TDM pipeline prologue: blocks 0 and 1 in flight.
  if (w == 0) {
    tdm_load_v(v_b + 0,  ldsv0);
    tdm_load_v(v_b + 32, ldsv1);
  }

  for (int blk = 0; blk < NSEQ / 32; ++blk) {
    const int m0  = blk * 32;
    const int cur = blk & 1;

    // Retire the oldest TDM load (in-order per wave): buffer `cur` is ready.
    if (w == 0) {
      if (blk < NSEQ / 32 - 1) __builtin_amdgcn_s_wait_tensorcnt(1);
      else                     __builtin_amdgcn_s_wait_tensorcnt(0);
    }
    __syncthreads();   // publish buffer `cur` to all 8 waves

    // --- S = q . k  (K tiles direct from L2-resident kT) ---
    const v16bf bk0 = *(const v16bf*)(kT_b + (size_t)(m0 + colN) * DCH + kb16);
    const v16bf bk1 = *(const v16bf*)(kT_b + (size_t)(m0 + 16 + colN) * DCH + kb16);
    const v8f zf = {};
    v8f S0 = wmma_bf16(a_q, bk0, zf);
    v8f S1 = wmma_bf16(a_q, bk1, zf);

    // --- online softmax ---
    float sc[8];
    v8f P0, P1;
#pragma unroll
    for (int r = 0; r < 8; ++r) {
      float rowmax = half_reduce_max(fmaxf(S0[r], S1[r]));
      float mnew = fmaxf(m_r[r], rowmax);
      sc[r] = __expf(m_r[r] - mnew);
      m_r[r] = mnew;
      P0[r] = __expf(S0[r] - mnew);
      P1[r] = __expf(S1[r] - mnew);
      float rsum = half_reduce_add(P0[r] + P1[r]);
      l_r[r] = l_r[r] * sc[r] + rsum;
    }
#pragma unroll
    for (int t = 0; t < 16; ++t) {
#pragma unroll
      for (int r = 0; r < 8; ++r) O[t][r] *= sc[r];
    }

    // --- P: C/D layout -> A layout via per-wave LDS tile ---
#pragma unroll
    for (int r = 0; r < 8; ++r) {
      const int row = (half << 3) + r;
      Plds[w][row][colN]      = (__bf16)P0[r];
      Plds[w][row][colN + 16] = (__bf16)P1[r];
    }
    __builtin_amdgcn_wave_barrier();
    const v16bf pa = load_a_tile(&Plds[w][0][0], 32, lane);
    __builtin_amdgcn_wave_barrier();

    // --- O += P . V^T, V block from TDM-staged LDS (80 B padded rows) ---
#pragma unroll
    for (int t = 0; t < 16; ++t) {
      const v16bf bv16 =
          *(const v16bf*)(Vlds[cur] + (t * 16 + colN) * VROW + (kb16 << 1));
      O[t] = wmma_bf16(pa, bv16, O[t]);
    }

    __syncthreads();   // all waves done with buffer `cur` before TDM reuses it
    if (w == 0 && blk + 2 < NSEQ / 32)
      tdm_load_v(v_b + (blk + 2) * 32, cur ? ldsv1 : ldsv0);
  }

  // --- epilogue: out = gamma * O / l + x ---
  float linv[8];
#pragma unroll
  for (int r = 0; r < 8; ++r) linv[r] = 1.0f / l_r[r];
  const int q0 = qbase + (half << 3);
#pragma unroll
  for (int t = 0; t < 16; ++t) {
    const int e = t * 16 + colN;
    const float* xp = x_b + (size_t)e * NSEQ + q0;
    float* op = o_b + (size_t)e * NSEQ + q0;
    const float4 x0 = *(const float4*)xp;
    const float4 x1 = *(const float4*)(xp + 4);
    float4 o0, o1;
    o0.x = g * O[t][0] * linv[0] + x0.x;
    o0.y = g * O[t][1] * linv[1] + x0.y;
    o0.z = g * O[t][2] * linv[2] + x0.z;
    o0.w = g * O[t][3] * linv[3] + x0.w;
    o1.x = g * O[t][4] * linv[4] + x1.x;
    o1.y = g * O[t][5] * linv[5] + x1.y;
    o1.z = g * O[t][6] * linv[6] + x1.z;
    o1.w = g * O[t][7] * linv[7] + x1.w;
    *(float4*)op = o0;
    *(float4*)(op + 4) = o1;
  }
}

// ---------------------------------------------------------------------------
extern "C" void kernel_launch(void* const* d_in, const int* in_sizes, int n_in,
                              void* d_out, int out_size, void* d_ws, size_t ws_size,
                              hipStream_t stream) {
  (void)in_sizes; (void)n_in; (void)out_size; (void)ws_size;
  const float* x     = (const float*)d_in[0];
  const float* Wq    = (const float*)d_in[1];
  const float* bq    = (const float*)d_in[2];
  const float* Wk    = (const float*)d_in[3];
  const float* bk    = (const float*)d_in[4];
  const float* Wv    = (const float*)d_in[5];
  const float* bv    = (const float*)d_in[6];
  const float* gamma = (const float*)d_in[7];
  float* out = (float*)d_out;

  char* ws = (char*)d_ws;
  __bf16* xT  = (__bf16*)(ws);                         // 16 MB
  __bf16* qT  = (__bf16*)(ws + 16777216);              //  2 MB
  __bf16* kT  = (__bf16*)(ws + 18874368);              //  2 MB
  __bf16* vb  = (__bf16*)(ws + 20971520);              // 16 MB
  __bf16* Wqb = (__bf16*)(ws + 37748736);              // 16 KB
  __bf16* Wkb = (__bf16*)(ws + 37748736 + 16384);      // 16 KB
  __bf16* Wvb = (__bf16*)(ws + 37748736 + 32768);      // 128 KB

  convert_w_kernel<<<256, 256, 0, stream>>>(Wq, Wk, Wv, Wqb, Wkb, Wvb);
  transpose_conv_kernel<<<dim3(NSEQ / 32, CCH / 32, BATCH), dim3(32, 8), 0, stream>>>(x, xT);
  proj_qk_kernel<<<1024, 256, 0, stream>>>(xT, Wqb, Wkb, bq, bk, qT, kT);
  proj_v_kernel<<<4096, 256, 0, stream>>>(xT, Wvb, bv, vb);
  attn_kernel<<<dim3(NSEQ / 16 / 8, BATCH), 256, 0, stream>>>(qT, kT, vb, x, gamma, out);
}
